// NSAttentionCausalDiscovery_55903294324923
// MI455X (gfx1250) — matrix operand order repaired and checked by reading, not compiled, not run on hardware
//
#include <hip/hip_runtime.h>
#include <hip/hip_bf16.h>
#include <stdint.h>

typedef __attribute__((ext_vector_type(16))) _Float16 v16h;
typedef __attribute__((ext_vector_type(8)))  float    v8f;

#define B_ 2
#define S_ 512
#define F_ 128
#define H_ 4
#define D_ 32

// ---------------------------------------------------------------------------
// CDNA5 async global->LDS copy (16 B per lane), tracked by ASYNCcnt.
// Probe-derived prototype: param0 is AS(1) pointer to int-vector(16B),
// param1 is the LDS destination, then imm offset + cpol.
// Falls back to synchronous VGPR round-trip copies if the builtin is absent.
// ---------------------------------------------------------------------------
#if defined(__HIP_DEVICE_COMPILE__) && __has_builtin(__builtin_amdgcn_global_load_async_to_lds_b128)
#define HAS_ASYNC_LDS 1
typedef int v4i_vs __attribute__((vector_size(16)));
typedef __attribute__((address_space(1))) v4i_vs* gptr16;
typedef __attribute__((address_space(3))) v4i_vs* lptr16;
__device__ __forceinline__ void async_copy16(const void* g, void* l) {
  __builtin_amdgcn_global_load_async_to_lds_b128(
      (gptr16)(uintptr_t)g, (lptr16)(uint32_t)(uintptr_t)l,
      /*offset=*/0, /*cpol=*/0);
}
__device__ __forceinline__ void async_wait0() {
#if __has_builtin(__builtin_amdgcn_s_wait_asynccnt)
  __builtin_amdgcn_s_wait_asynccnt(0);
#else
  asm volatile("s_wait_asynccnt 0x0" ::: "memory");
#endif
}
#else
#define HAS_ASYNC_LDS 0
#endif

// ---------------------------------------------------------------------------
// Kernel 1: projections.  aP = X@W1[:F] + b1 (b1 folded), cP = X@W1[F:],
// qP = X@Wq + bq, kP = X@Wk + bk.  One block per (b,s) row, 128 threads.
// ---------------------------------------------------------------------------
__global__ __launch_bounds__(128)
void proj_kernel(const float* __restrict__ X,
                 const float* __restrict__ Wq, const float* __restrict__ bq,
                 const float* __restrict__ Wk, const float* __restrict__ bk,
                 const float* __restrict__ W1, const float* __restrict__ b1,
                 float* __restrict__ aP, float* __restrict__ cP,
                 float* __restrict__ qP, float* __restrict__ kP) {
  __shared__ float xrow[F_];
  const int row = blockIdx.x;        // b*S + s
  const int n   = threadIdx.x;       // 0..127
  xrow[n] = X[row * F_ + n];
  __syncthreads();
  float sa = 0.f, sc = 0.f, sq = 0.f, sk = 0.f;
#pragma unroll 4
  for (int k2 = 0; k2 < F_; ++k2) {
    const float x = xrow[k2];
    sa += x * W1[k2 * F_ + n];
    sc += x * W1[(F_ + k2) * F_ + n];
    sq += x * Wq[k2 * F_ + n];
    sk += x * Wk[k2 * F_ + n];
  }
  aP[row * F_ + n] = sa + b1[n];
  cP[row * F_ + n] = sc;
  qP[row * F_ + n] = sq + bq[n];
  kP[row * F_ + n] = sk + bk[n];
}

// ---------------------------------------------------------------------------
// Kernel 2: repack W2 [128,64] f32 -> f16 B-fragments for
// v_wmma_f32_16x16x32_f16 (wave32).  Fragment (nt,ks): lane L holds
// column n = nt*16 + (L&15); 16 halves = K = ks*32 + 16*(L>>4) + h.
// ---------------------------------------------------------------------------
__global__ __launch_bounds__(512)
void pack_w2_kernel(const float* __restrict__ W2, _Float16* __restrict__ w2fG) {
  const int t  = threadIdx.x;     // 0..511 == one fragment-lane
  const int L  = t & 31;
  const int ks = (t >> 5) & 3;
  const int nt = t >> 7;
  const int n  = nt * 16 + (L & 15);
  const int g  = L >> 4;
#pragma unroll
  for (int h = 0; h < 16; ++h) {
    const int K = ks * 32 + g * 16 + h;
    w2fG[((nt * 4 + ks) * 32 + L) * 16 + h] = (_Float16)W2[K * 64 + n];
  }
}

// ---------------------------------------------------------------------------
// Kernel 3: attention.  One block per (b,i), 4 waves = 4 heads.
// scores -> mask(-inf) -> softmax (wave shuffles) -> mean over heads.
// ---------------------------------------------------------------------------
__global__ __launch_bounds__(128)
void attn_kernel(const float* __restrict__ qP, const float* __restrict__ kP,
                 const unsigned char* __restrict__ mask,
                 float* __restrict__ attnW) {
  __shared__ float rows[H_][S_];
  const int b    = blockIdx.x / S_;
  const int i    = blockIdx.x % S_;
  const int wave = threadIdx.x >> 5;   // head
  const int lane = threadIdx.x & 31;
  const float scale = 0.17677669529663687f;  // 1/sqrt(32)

  float qv[D_];
  const float* qrow = qP + (b * S_ + i) * F_ + wave * D_;
#pragma unroll
  for (int d = 0; d < D_; ++d) qv[d] = qrow[d];

  float sc[16];
#pragma unroll
  for (int t = 0; t < 16; ++t) {
    const int j = lane + 32 * t;
    const float* krow = kP + (b * S_ + j) * F_ + wave * D_;
    float s = 0.f;
#pragma unroll
    for (int d = 0; d < D_; ++d) s += qv[d] * krow[d];
    sc[t] = mask[i * S_ + j] ? s * scale : -__builtin_inff();
  }
  float mx = sc[0];
#pragma unroll
  for (int t = 1; t < 16; ++t) mx = fmaxf(mx, sc[t]);
#pragma unroll
  for (int off = 16; off > 0; off >>= 1) mx = fmaxf(mx, __shfl_xor(mx, off, 32));
  float sum = 0.f;
#pragma unroll
  for (int t = 0; t < 16; ++t) { sc[t] = __expf(sc[t] - mx); sum += sc[t]; }
#pragma unroll
  for (int off = 16; off > 0; off >>= 1) sum += __shfl_xor(sum, off, 32);
  const float inv = 1.f / sum;
#pragma unroll
  for (int t = 0; t < 16; ++t) rows[wave][lane + 32 * t] = sc[t] * inv;
  __syncthreads();
  for (int j = threadIdx.x; j < S_; j += 128)
    attnW[((size_t)(b * S_ + i)) * S_ + j] =
        0.25f * (rows[0][j] + rows[1][j] + rows[2][j] + rows[3][j]);
}

// ---------------------------------------------------------------------------
// Kernel 4 (hot): fused edge MLP.  256 threads = 8 waves; each wave owns
// (i = ig*8+wave, j-tile of 16).  16x v_wmma_f32_16x16x32_f16 per wave.
// h1 is built in registers from LDS-staged a/c rows, never materialized.
// LDS staging uses the CDNA5 async global->LDS path when available.
// ---------------------------------------------------------------------------
__global__ __launch_bounds__(256)
void edge_kernel(const float* __restrict__ aP, const float* __restrict__ cP,
                 const _Float16* __restrict__ w2fG,
                 const float* __restrict__ W3, const float* __restrict__ b3,
                 const float* __restrict__ b2,
                 const float* __restrict__ attnW, float* __restrict__ out) {
  __shared__ __align__(16) _Float16 w2f[4 * 4 * 32 * 16];  // 16 KB fragments
  __shared__ __align__(16) float ctile[16][F_ + 4];        // +4 pad: no bank conflicts
  __shared__ __align__(16) float atile[8][F_];

  const int bi  = blockIdx.x;
  const int b   = bi / (32 * 64);
  const int rem = bi % (32 * 64);
  const int jt  = rem / 64;
  const int ig  = rem % 64;
  const int j0  = jt * 16;
  const int tid = threadIdx.x;

#if HAS_ASYNC_LDS
  // --- async global->LDS staging (ASYNCcnt path, no VGPR round-trip) ---
  {
    const uint4* src = (const uint4*)w2fG;
    uint4*       dst = (uint4*)w2f;
    for (int t = tid; t < 1024; t += 256)           // 16 KB of W2 fragments
      async_copy16(src + t, dst + t);
  }
  for (int seg = tid; seg < 512; seg += 256) {      // 16 c-rows (16 B segs)
    const int r = seg >> 5, c = (seg & 31) * 4;
    async_copy16(&cP[(b * S_ + j0 + r) * F_ + c], &ctile[r][c]);
  }
  {                                                 // 8 a-rows (one seg/thread)
    const int r = tid >> 5, c = (tid & 31) * 4;
    async_copy16(&aP[(b * S_ + ig * 8 + r) * F_ + c], &atile[r][c]);
  }
  async_wait0();           // own async copies done …
#else
  // --- fallback: synchronous cooperative copies ---
  {
    const uint4* src = (const uint4*)w2fG;
    uint4*       dst = (uint4*)w2f;
    for (int t = tid; t < 1024; t += 256) dst[t] = src[t];
  }
  for (int t = tid; t < 16 * F_; t += 256) {
    const int r = t >> 7, c = t & 127;
    ctile[r][c] = cP[(b * S_ + j0 + r) * F_ + c];
  }
  for (int t = tid; t < 8 * F_; t += 256) {
    const int r = t >> 7, c = t & 127;
    atile[r][c] = aP[(b * S_ + ig * 8 + r) * F_ + c];
  }
#endif
  __syncthreads();         // … and every other wave's too

  const int wave = tid >> 5, lane = tid & 31;
  const int i = ig * 8 + wave;
  const int m = lane & 15;     // A row (j index) / C column (mod 16)
  const int g = lane >> 4;     // lane half
  const float* arow = atile[wave];
  const float* crow = ctile[m];

  // Build the 4 A-fragments (K slices of 32): h1 = relu(a_i + c_j) as f16.
  // A k order used here (contiguous; pack_w2 uses the same): lane L -> row M=L%16; half t -> K =
  // ks*32 + 16*(L/16) + t.
  v16h A[4];
#pragma unroll
  for (int ks = 0; ks < 4; ++ks) {
#pragma unroll
    for (int t = 0; t < 16; ++t) {
      const int K = ks * 32 + g * 16 + t;
      const float v = arow[K] + crow[K];
      A[ks][t] = (_Float16)fmaxf(v, 0.f);
    }
  }

  const v16h* Bf = (const v16h*)w2f;
  float part[8];
#pragma unroll
  for (int v = 0; v < 8; ++v) part[v] = 0.f;
  const float b3v = b3[0];

#pragma unroll
  for (int nt = 0; nt < 4; ++nt) {
    const int n = nt * 16 + m;          // C/D layout: lane holds column n
    const float b2n = b2[n];
    v8f acc;
#pragma unroll
    for (int v = 0; v < 8; ++v) acc[v] = b2n;   // fold +b2 into C
#pragma unroll
    for (int ks = 0; ks < 4; ++ks) {
      const v16h Bv = Bf[(nt * 4 + ks) * 32 + lane];
      acc = __builtin_amdgcn_wmma_f32_16x16x32_f16(
          /*neg_a=*/false, A[ks], /*neg_b=*/false, Bv,
          /*c_mod=*/(short)0, acc, /*reuse_a=*/false, /*reuse_b=*/false);
    }
    const float w3n = W3[n];
#pragma unroll
    for (int v = 0; v < 8; ++v) part[v] += fmaxf(acc[v], 0.f) * w3n;
  }

  // Reduce the 64 N-partials: sum across the 16 lanes of each half.
#pragma unroll
  for (int off = 1; off < 16; off <<= 1) {
#pragma unroll
    for (int v = 0; v < 8; ++v) part[v] += __shfl_xor(part[v], off, 32);
  }

  if (m == 0) {  // lanes 0 (rows 0..7) and 16 (rows 8..15) write results
    const int m0 = g * 8;
    const float* awrow = attnW + ((size_t)(b * S_ + i)) * S_;
    float*       orow  = out   + ((size_t)(b * S_ + i)) * S_;
#pragma unroll
    for (int v = 0; v < 8; ++v) {
      const int j = j0 + m0 + v;
      float e = 1.f / (1.f + __expf(-(part[v] + b3v)));
      if (j == i) e = 0.f;                 // zero diagonal
      orow[j] = e * awrow[j];
    }
  }
}

// ---------------------------------------------------------------------------
extern "C" void kernel_launch(void* const* d_in, const int* in_sizes, int n_in,
                              void* d_out, int out_size, void* d_ws, size_t ws_size,
                              hipStream_t stream) {
  (void)in_sizes; (void)n_in; (void)out_size; (void)ws_size;
  const float*         X    = (const float*)d_in[0];
  const unsigned char* mask = (const unsigned char*)d_in[1];   // jnp bool -> 1 byte
  const float* Wq = (const float*)d_in[2];
  const float* bq = (const float*)d_in[3];
  const float* Wk = (const float*)d_in[4];
  const float* bk = (const float*)d_in[5];
  const float* W1 = (const float*)d_in[6];
  const float* b1 = (const float*)d_in[7];
  const float* W2 = (const float*)d_in[8];
  const float* b2 = (const float*)d_in[9];
  const float* W3 = (const float*)d_in[10];
  const float* b3 = (const float*)d_in[11];
  float* out = (float*)d_out;

  // Workspace layout (bytes)
  char* ws = (char*)d_ws;
  float*    attnW = (float*)(ws + 0);                  // 2*512*512 f32 = 2 MB
  float*    aP    = (float*)(ws + 2097152);            // 512 KB
  float*    cP    = (float*)(ws + 2621440);            // 512 KB
  float*    qP    = (float*)(ws + 3145728);            // 512 KB
  float*    kP    = (float*)(ws + 3670016);            // 512 KB
  _Float16* w2fG  = (_Float16*)(ws + 4194304);         // 16 KB

  proj_kernel<<<B_ * S_, 128, 0, stream>>>(X, Wq, bq, Wk, bk, W1, b1,
                                           aP, cP, qP, kP);
  pack_w2_kernel<<<1, 512, 0, stream>>>(W2, w2fG);
  attn_kernel<<<B_ * S_, 128, 0, stream>>>(qP, kP, mask, attnW);
  edge_kernel<<<B_ * (S_ / 16) * (S_ / 8), 256, 0, stream>>>(
      aP, cP, w2fG, W3, b3, b2, attnW, out);
}
